// MultiScaleLDSA_ChunkedInference_20100446945645
// MI455X (gfx1250) — matrix operand order repaired and checked by reading, not compiled, not run on hardware
//
#include <hip/hip_runtime.h>
#include <hip/hip_bf16.h>

// ---------------------------------------------------------------------------
// MultiScale local-window attention for MI455X (gfx1250, wave32, WMMA).
//
// Pipeline (all GEMMs via v_wmma_f32_16x16x32_f16, f32 accumulate):
//   0. convert all weights f32 -> f16 (one-time, tiny)
//   per scale s in {0,1}:
//     1. q_s   = relu(query @ W1_s)   -> f16   (relu fused in epilogue)
//     2. v3_s  = value @ W3_s         -> f32
//     3. lg_s  = q_s @ W2_s           -> f32   (A staged via async-to-LDS)
//     4. softmax over ctx + sliding-window weighted sum of v3_s,
//        accumulated with softmax(scale_w)[s]  (wave-per-(t,head), shuffles)
//   5. out = acc @ w_out -> f32 d_out
//
// GEMM structure: 128x128 block tile, BK=32, software-pipelined DOUBLE
// BUFFER: global loads (or async global->LDS copies) for tile t+1 are issued
// before the WMMAs on tile t, so memory latency hides under matrix math.
// B tiles staged TRANSPOSED in LDS so every CDNA5 B-fragment is two contiguous
// 16-byte runs => two ds_load_b128 per fragment.
// ---------------------------------------------------------------------------

typedef __attribute__((ext_vector_type(16))) _Float16 v16h;
typedef __attribute__((ext_vector_type(8)))  _Float16 v8h;
typedef __attribute__((ext_vector_type(8)))  float    v8f;
typedef __attribute__((ext_vector_type(4)))  _Float16 v4h;

#define NFEAT   256
#define TDIM    4096
#define BTOT    8192      // B*T
#define HPSC    4         // heads per scale
#define HDIM    64        // NFEAT / HPSC

// -------------------------- weight conversion ------------------------------
__global__ void cvt_f32_to_f16(const float* __restrict__ src,
                               _Float16* __restrict__ dst, int n) {
  int i = blockIdx.x * blockDim.x + threadIdx.x;
  if (i < n) dst[i] = (_Float16)src[i];
}

// --------------------- async global->LDS (gfx1250) -------------------------
// Per-lane 16B copy. LDS destination offset = low 32 bits of the generic
// pointer (AMDGPU generic LDS pointers are {aperture_hi, lds_offset}).
__device__ __forceinline__ void async_copy_b128(void* lds_dst, const void* gsrc) {
  unsigned lds_off = (unsigned)(size_t)lds_dst;
  asm volatile("global_load_async_to_lds_b128 %0, %1, off"
               :: "v"(lds_off), "v"(gsrc) : "memory");
}
__device__ __forceinline__ void async_wait0() {
  asm volatile("s_wait_asynccnt 0x0" ::: "memory");
}

// -------------------------- WMMA tiled GEMM --------------------------------
// C[M,N] = op(A[M,K] @ B[K,N]);  A: f32 or f16 (A_HALF), B: f16, C: f32/f16.
// 256 threads = 8 waves; each wave owns a 64x32 macro-tile = 4x2 WMMA tiles.
// M % 128 == 0, K % 32 == 0, N % 4 == 0 guaranteed by the caller.
template<bool A_HALF, bool RELU, bool OUT_HALF>
__global__ __launch_bounds__(256)
void gemm_wmma(const void* __restrict__ Aptr, const _Float16* __restrict__ B,
               void* __restrict__ Cptr, int M, int N, int K, int nTilesN)
{
  constexpr int BM = 128, BN = 128, BK = 32;
  __shared__ _Float16 As [2][BM][BK + 8];   // row-major  [m][k], 80B stride
  __shared__ _Float16 BsT[2][BN][BK + 8];   // TRANSPOSED [n][k], 80B stride

  const int tid  = threadIdx.x;
  const int wid  = tid >> 5;
  const int lane = tid & 31;
  const int m0 = (blockIdx.x / nTilesN) * BM;
  const int n0 = (blockIdx.x % nTilesN) * BN;
  const int wm = (wid >> 2) * 64;       // wave row offset: 0 / 64
  const int wn = (wid & 3) * 32;        // wave col offset: 0..96
  const int r  = lane & 15;
  const int kh = lane >> 4;             // 0/1: half-wave selector

  // ---- staging helpers (loads batched separately from LDS stores) ----
  const int krow0   = (tid >> 5) * 4;   // B: 4 k-rows per thread
  const int c4      = (tid & 31) * 4;   // B: 4-column group
  const int colbase = n0 + c4;
  const bool binb   = colbase < N;      // N % 4 == 0 => group-uniform bound

  v4h   brow[4];                        // B next-tile registers
  float4 areg[4];                       // A next-tile registers (f32 path)

  auto loadBtile = [&](int kt) {
    #pragma unroll
    for (int rr = 0; rr < 4; ++rr) {
      if (binb) {
        brow[rr] = *(const v4h*)(B + (size_t)(kt + krow0 + rr) * N + colbase);
      } else {
        v4h z = {(_Float16)0.0f, (_Float16)0.0f, (_Float16)0.0f, (_Float16)0.0f};
        brow[rr] = z;
      }
    }
  };
  auto storeBtile = [&](int buf) {      // 4x4 register transpose -> b64 stores
    #pragma unroll
    for (int j = 0; j < 4; ++j) {
      v4h w = { brow[0][j], brow[1][j], brow[2][j], brow[3][j] };
      *(v4h*)&BsT[buf][c4 + j][krow0] = w;
    }
  };
  auto loadAtile = [&](int kt) {        // f32 path: 4x float4 loads
    const float* A = (const float*)Aptr;
    #pragma unroll
    for (int p = 0; p < 4; ++p) {
      int row = (tid >> 3) + p * 32;
      int k4  = (tid & 7) * 4;
      areg[p] = *(const float4*)(A + (size_t)(m0 + row) * K + kt + k4);
    }
  };
  auto storeAtile = [&](int buf) {      // cvt f32->f16 + b64 stores
    #pragma unroll
    for (int p = 0; p < 4; ++p) {
      int row = (tid >> 3) + p * 32;
      int k4  = (tid & 7) * 4;
      v4h hv = { (_Float16)areg[p].x, (_Float16)areg[p].y,
                 (_Float16)areg[p].z, (_Float16)areg[p].w };
      *(v4h*)&As[buf][row][k4] = hv;
    }
  };
  auto asyncAtile = [&](int kt, int buf) {  // f16 path: async global->LDS
    const _Float16* A = (const _Float16*)Aptr;
    #pragma unroll
    for (int p = 0; p < 2; ++p) {
      int row = (tid >> 2) + p * 64;
      int seg = (tid & 3) * 8;
      async_copy_b128(&As[buf][row][seg],
                      A + (size_t)(m0 + row) * K + kt + seg);
    }
  };

  v8f acc[4][2];
  #pragma unroll
  for (int i = 0; i < 4; ++i)
    #pragma unroll
    for (int j = 0; j < 2; ++j) acc[i][j] = {};

  // ---- prologue: stage tile 0 into buffer 0 ----
  if (A_HALF) asyncAtile(0, 0);
  else        loadAtile(0);
  loadBtile(0);
  if (!A_HALF) storeAtile(0);
  storeBtile(0);
  if (A_HALF) async_wait0();
  __syncthreads();

  const int NK = K / BK;
  for (int ti = 0; ti < NK; ++ti) {
    const int cur = ti & 1, nxt = cur ^ 1;
    const bool hn = (ti + 1) < NK;

    // issue next tile's global traffic BEFORE this tile's math
    if (hn) {
      if (A_HALF) asyncAtile((ti + 1) * BK, nxt);  // async engine overlaps WMMA
      else        loadAtile((ti + 1) * BK);
      loadBtile((ti + 1) * BK);
    }

    // ---- fragments (contiguous 16B runs => ds_load_b128) + 8 WMMAs ----
    v16h afrag[4], bfrag[2];
    #pragma unroll
    for (int tm = 0; tm < 4; ++tm) {
      int arow = wm + tm * 16 + r;
      v8h lo = *(const v8h*)&As[cur][arow][kh * 8];
      v8h hi = *(const v8h*)&As[cur][arow][16 + kh * 8];
      afrag[tm] = __builtin_shufflevector(lo, hi, 0,1,2,3,4,5,6,7,
                                                  8,9,10,11,12,13,14,15);
    }
    #pragma unroll
    for (int tn = 0; tn < 2; ++tn) {
      int bcol = wn + tn * 16 + r;
      v8h lo = *(const v8h*)&BsT[cur][bcol][kh * 8];
      v8h hi = *(const v8h*)&BsT[cur][bcol][16 + kh * 8];
      bfrag[tn] = __builtin_shufflevector(lo, hi, 0,1,2,3,4,5,6,7,
                                                  8,9,10,11,12,13,14,15);
    }
    #pragma unroll
    for (int tm = 0; tm < 4; ++tm)
      #pragma unroll
      for (int tn = 0; tn < 2; ++tn)
        acc[tm][tn] = __builtin_amdgcn_wmma_f32_16x16x32_f16(
            false, afrag[tm], false, bfrag[tn],
            (short)0, acc[tm][tn], false, false);

    // drain next tile into the other LDS buffer, then one barrier
    if (hn) {
      if (!A_HALF) storeAtile(nxt);
      storeBtile(nxt);
      if (A_HALF) async_wait0();
    }
    __syncthreads();
  }

  // ---- epilogue: C/D layout => VGPR e holds M = e + kh*8, N = lane&15 ----
  #pragma unroll
  for (int tm = 0; tm < 4; ++tm) {
    int cm = m0 + wm + tm * 16 + kh * 8;
    #pragma unroll
    for (int tn = 0; tn < 2; ++tn) {
      int cn = n0 + wn + tn * 16 + r;
      if (cn < N) {
        #pragma unroll
        for (int e = 0; e < 8; ++e) {
          float v = acc[tm][tn][e];
          if (RELU) v = v > 0.0f ? v : 0.0f;
          size_t off = (size_t)(cm + e) * N + cn;
          if (OUT_HALF) ((_Float16*)Cptr)[off] = (_Float16)v;
          else          ((float*)Cptr)[off] = v;
        }
      }
    }
  }
}

// ------------------- fused softmax + sliding-window gather -----------------
// One wave per (b,t,head). ctx <= 61 -> 2 logits per lane; max/sum by
// wave32 shuffles; probs broadcast via LDS; each lane owns 2 of 64 head dims.
template<int CTX, bool ACCUM>
__global__ __launch_bounds__(256)
void window_softmax(const float* __restrict__ logits,
                    const float* __restrict__ v3,
                    float* __restrict__ accout,
                    const float* __restrict__ scale_w, int sidx)
{
  __shared__ float probs[8][64];
  const int tid  = threadIdx.x;
  const int wid  = tid >> 5;
  const int lane = tid & 31;
  const int g  = blockIdx.x * 8 + wid;   // 0..32767  (BTOT*HPSC waves)
  const int bt = g >> 2;                 // 0..8191
  const int h  = g & 3;
  const int t  = bt & (TDIM - 1);
  constexpr int HALF = (CTX - 1) / 2;

  const float* lrow = logits + (size_t)bt * (HPSC * CTX) + (size_t)h * CTX;
  float l0 = (lane      < CTX) ? lrow[lane]      : -3.0e38f;
  float l1 = (lane + 32 < CTX) ? lrow[lane + 32] : -3.0e38f;
  float m = fmaxf(l0, l1);
  #pragma unroll
  for (int o = 16; o > 0; o >>= 1) m = fmaxf(m, __shfl_xor(m, o, 32));
  float e0 = (lane      < CTX) ? __expf(l0 - m) : 0.0f;
  float e1 = (lane + 32 < CTX) ? __expf(l1 - m) : 0.0f;
  float s = e0 + e1;
  #pragma unroll
  for (int o = 16; o > 0; o >>= 1) s += __shfl_xor(s, o, 32);
  float inv = 1.0f / s;
  probs[wid][lane]      = e0 * inv;
  probs[wid][lane + 32] = e1 * inv;
  __syncthreads();

  // softmax over the 2 scale weights (uniform scalar work)
  float a0 = scale_w[0], a1 = scale_w[1];
  float mm = fmaxf(a0, a1);
  float x0 = __expf(a0 - mm), x1 = __expf(a1 - mm);
  float sw = ((sidx == 0) ? x0 : x1) / (x0 + x1);

  const int bbase = bt - t;              // = b * TDIM
  float2 out = {0.0f, 0.0f};
  for (int c = 0; c < CTX; ++c) {
    int tt = t - HALF + c;
    if (tt < 0 || tt >= TDIM) continue;  // zero padding of the window
    float p = probs[wid][c];             // LDS broadcast (uniform per wave)
    const float2* vr = (const float2*)(v3 + (size_t)(bbase + tt) * NFEAT
                                          + (size_t)h * HDIM);
    float2 vv = vr[lane];
    out.x += p * vv.x;
    out.y += p * vv.y;
  }

  float2* optr = (float2*)(accout + (size_t)bt * NFEAT + (size_t)h * HDIM);
  if (ACCUM) {
    float2 prev = optr[lane];
    prev.x += sw * out.x;
    prev.y += sw * out.y;
    optr[lane] = prev;
  } else {
    float2 w = { sw * out.x, sw * out.y };
    optr[lane] = w;
  }
}

// ------------------------------ launcher -----------------------------------
extern "C" void kernel_launch(void* const* d_in, const int* in_sizes, int n_in,
                              void* d_out, int out_size, void* d_ws, size_t ws_size,
                              hipStream_t stream) {
  (void)in_sizes; (void)n_in; (void)out_size; (void)ws_size;
  const float* query   = (const float*)d_in[0];
  /* d_in[1] = key is unused by the reference */
  const float* value   = (const float*)d_in[2];
  const float* w1f[2]  = {(const float*)d_in[3], (const float*)d_in[4]};
  const float* w2f[2]  = {(const float*)d_in[5], (const float*)d_in[6]};
  const float* w3f[2]  = {(const float*)d_in[7], (const float*)d_in[8]};
  const float* scale_w = (const float*)d_in[9];
  const float* woutf   = (const float*)d_in[10];

  const int CTXS[2] = {31, 61};

  // ---- carve workspace ----
  char* ws = (char*)d_ws;
  size_t off = 0;
  auto carve = [&](size_t bytes) -> void* {
    void* p = ws + off;
    off += bytes;
    off = (off + 255) & ~(size_t)255;
    return p;
  };
  _Float16* w1h[2]; _Float16* w2h[2]; _Float16* w3h[2];
  w1h[0] = (_Float16*)carve(NFEAT * NFEAT * sizeof(_Float16));
  w1h[1] = (_Float16*)carve(NFEAT * NFEAT * sizeof(_Float16));
  w3h[0] = (_Float16*)carve(NFEAT * NFEAT * sizeof(_Float16));
  w3h[1] = (_Float16*)carve(NFEAT * NFEAT * sizeof(_Float16));
  _Float16* wouth = (_Float16*)carve(NFEAT * NFEAT * sizeof(_Float16));
  w2h[0] = (_Float16*)carve((size_t)NFEAT * HPSC * CTXS[0] * sizeof(_Float16));
  w2h[1] = (_Float16*)carve((size_t)NFEAT * HPSC * CTXS[1] * sizeof(_Float16));
  _Float16* qh   = (_Float16*)carve((size_t)BTOT * NFEAT * sizeof(_Float16));
  float* v3f     = (float*)carve((size_t)BTOT * NFEAT * sizeof(float));
  float* lg      = (float*)carve((size_t)BTOT * HPSC * CTXS[1] * sizeof(float));
  float* accb    = (float*)carve((size_t)BTOT * NFEAT * sizeof(float));

  // ---- 0: weight conversion ----
  const int CT = 256;
  cvt_f32_to_f16<<<(NFEAT*NFEAT + CT-1)/CT, CT, 0, stream>>>(w1f[0], w1h[0], NFEAT*NFEAT);
  cvt_f32_to_f16<<<(NFEAT*NFEAT + CT-1)/CT, CT, 0, stream>>>(w1f[1], w1h[1], NFEAT*NFEAT);
  cvt_f32_to_f16<<<(NFEAT*NFEAT + CT-1)/CT, CT, 0, stream>>>(w3f[0], w3h[0], NFEAT*NFEAT);
  cvt_f32_to_f16<<<(NFEAT*NFEAT + CT-1)/CT, CT, 0, stream>>>(w3f[1], w3h[1], NFEAT*NFEAT);
  cvt_f32_to_f16<<<(NFEAT*NFEAT + CT-1)/CT, CT, 0, stream>>>(woutf, wouth, NFEAT*NFEAT);
  { int n0 = NFEAT * HPSC * CTXS[0];
    cvt_f32_to_f16<<<(n0 + CT-1)/CT, CT, 0, stream>>>(w2f[0], w2h[0], n0); }
  { int n1 = NFEAT * HPSC * CTXS[1];
    cvt_f32_to_f16<<<(n1 + CT-1)/CT, CT, 0, stream>>>(w2f[1], w2h[1], n1); }

  // ---- per-scale pipeline ----
  const int grid256 = (BTOT / 128) * 2;   // 128 blocks for N=256
  for (int s = 0; s < 2; ++s) {
    // q_s = relu(query @ W1_s) -> f16
    gemm_wmma<false, true, true><<<grid256, 256, 0, stream>>>(
        query, w1h[s], qh, BTOT, NFEAT, NFEAT, 2);
    // v3_s = value @ W3_s -> f32
    gemm_wmma<false, false, false><<<grid256, 256, 0, stream>>>(
        value, w3h[s], v3f, BTOT, NFEAT, NFEAT, 2);
    // logits = q_s @ W2_s -> f32 (N = 4*ctx), A staged via async-to-LDS
    int Ns  = HPSC * CTXS[s];
    int ntn = (Ns + 127) / 128;
    gemm_wmma<true, false, false><<<(BTOT / 128) * ntn, 256, 0, stream>>>(
        qh, w2h[s], lg, BTOT, Ns, NFEAT, ntn);
    // fused softmax + window gather, mixed by softmax(scale_w)[s]
    if (s == 0)
      window_softmax<31, false><<<(BTOT * HPSC) / 8, 256, 0, stream>>>(
          lg, v3f, accb, scale_w, 0);
    else
      window_softmax<61, true><<<(BTOT * HPSC) / 8, 256, 0, stream>>>(
          lg, v3f, accb, scale_w, 1);
  }

  // ---- 5: out = acc @ w_out -> f32 d_out ----
  gemm_wmma<false, false, false><<<grid256, 256, 0, stream>>>(
      accb, wouth, d_out, BTOT, NFEAT, NFEAT, 2);
}
